// TGAT_45466523795883
// MI455X (gfx1250) — matrix operand order repaired and checked by reading, not compiled, not run on hardware
//
#include <hip/hip_runtime.h>
#include <hip/hip_bf16.h>
#include <stdint.h>

// Problem constants (match reference)
#define BB 4096
#define NN 64
#define DD 128
#define MM 384   // D + De + Dt
#define HH 2
#define DK 192

#define LDSROW 132   // 128 dwords + 4 pad dwords (written by TDM pad feature)

typedef __attribute__((ext_vector_type(8)))  float  v8f;
typedef __attribute__((ext_vector_type(16))) __bf16 v16bf;
typedef __attribute__((ext_vector_type(4)))  unsigned int v4u;
typedef __attribute__((ext_vector_type(4)))  int    v4i;
typedef __attribute__((ext_vector_type(8)))  int    v8i;

union FragBF { v16bf v; uint4 u[2]; };

static __device__ inline v8f wmma_bf16(v16bf a, v16bf b, v8f c) {
  // v_wmma_f32_16x16x32_bf16: (neg_a, A, neg_b, B, c_mod, C, reuse_a, reuse_b)
  return __builtin_amdgcn_wmma_f32_16x16x32_bf16(false, a, false, b, (short)0, c,
                                                 false, false);
}

// ---------------- prep kernels ----------------

__global__ __launch_bounds__(256) void cvt_bf16_kernel(const float* __restrict__ s,
                                                       __bf16* __restrict__ d, int n) {
  int i = blockIdx.x * 256 + threadIdx.x;
  if (i < n) d[i] = (__bf16)s[i];
}

// d[m*384 + j] = s[j*384 + m]   (Wk^T, bf16)
__global__ __launch_bounds__(256) void transpose384_kernel(const float* __restrict__ s,
                                                           __bf16* __restrict__ d) {
  int i = blockIdx.x * 256 + threadIdx.x;
  if (i < MM * MM) {
    int m = i / MM, j = i - m * MM;
    d[i] = (__bf16)s[j * MM + m];
  }
}

// Q0 = [src | zeros(De) | src_t]  as bf16 (B x 384)
__global__ __launch_bounds__(256) void build_q0_kernel(const float* __restrict__ src,
                                                       const float* __restrict__ src_t,
                                                       __bf16* __restrict__ q0b) {
  int i = blockIdx.x * 256 + threadIdx.x;
  if (i < BB * MM) {
    int b = i / MM, j = i - b * MM;
    float v = 0.f;
    if (j < 128)       v = src[b * 128 + j];
    else if (j >= 256) v = src_t[b * 128 + (j - 256)];
    q0b[i] = (__bf16)v;
  }
}

// ---------------- generic bf16 WMMA GEMM ----------------
// C[row, col] = sum_k A[row, k] * W[col, k]  (+bias[col]) (optional relu)
// A: rows x K (row stride lda, bf16), W: cols x K (row stride ldw, bf16)
// Block = 256 threads (8 waves), computes a 64x64 output block.
// flags: bit0 = relu
__global__ __launch_bounds__(256) void gemm_bf16_kernel(
    const __bf16* __restrict__ A, int lda,
    const __bf16* __restrict__ W, int ldw,
    float* __restrict__ Cf, __bf16* __restrict__ Cb, int ldc,
    const float* __restrict__ bias, int K, int flags)
{
  const int lane = threadIdx.x & 31;
  const int wv   = threadIdx.x >> 5;   // 0..7
  const int rt   = wv & 3;             // row tile within block (0..3)
  const int cp   = wv >> 2;            // col pair (0..1), each covers 2 tiles
  const int row0 = blockIdx.y * 64 + rt * 16;
  const int colb = blockIdx.x * 64 + cp * 32;

  const int r   = lane & 15;
  const int hiA = (lane >> 4) << 3;    // A frag: K halves 0..7 / 8..15 (+16 for upper)
  const int hiB = (lane >> 4) << 3;    // B frag: K halves 0..7 / 8..15 (+16 for upper)

  const __bf16* Ap  = A + (size_t)(row0 + r) * lda;
  const __bf16* Wp0 = W + (size_t)(colb + r)      * ldw;
  const __bf16* Wp1 = W + (size_t)(colb + 16 + r) * ldw;

  v8f acc0 = {};
  v8f acc1 = {};

  for (int kk = 0; kk < K; kk += 32) {
    FragBF a, b0, b1;
    a.u[0]  = *(const uint4*)(Ap  + kk + hiA);        // K = hiA .. hiA+7
    a.u[1]  = *(const uint4*)(Ap  + kk + 16 + hiA);   // K = 16+hiA .. 16+hiA+7
    b0.u[0] = *(const uint4*)(Wp0 + kk + hiB);        // K = hiB .. hiB+7
    b0.u[1] = *(const uint4*)(Wp0 + kk + 16 + hiB);   // K = 16+hiB .. 16+hiB+7
    b1.u[0] = *(const uint4*)(Wp1 + kk + hiB);
    b1.u[1] = *(const uint4*)(Wp1 + kk + 16 + hiB);
    acc0 = wmma_bf16(a.v, b0.v, acc0);
    acc1 = wmma_bf16(a.v, b1.v, acc1);
  }

  const int hiC = (lane >> 4) << 3;    // C: m = v + 8*(lane>=16), n = lane&15
  #pragma unroll
  for (int t = 0; t < 2; ++t) {
    const int col = colb + t * 16 + (lane & 15);
    const float bsv = bias ? bias[col] : 0.f;
    v8f acc = t ? acc1 : acc0;
    #pragma unroll
    for (int v = 0; v < 8; ++v) {
      const int mrow = row0 + v + hiC;
      float val = acc[v] + bsv;
      if (flags & 1) val = fmaxf(val, 0.f);
      if (Cf) Cf[(size_t)mrow * ldc + col] = val;
      if (Cb) Cb[(size_t)mrow * ldc + col] = (__bf16)val;
    }
  }
}

// ---------------- TDM descriptor helper ----------------
// Issue one TDM load: 2-D tile (128 x 64 fp32) global -> LDS, with the TDM pad
// feature inserting 4 pad DWORDs after every 128 DWORDs (row stride 132 in LDS).
static __device__ inline void tdm_load_tile_64x128(const float* gptr,
                                                   unsigned int lds_byte_off) {
#if __has_builtin(__builtin_amdgcn_tensor_load_to_lds)
  unsigned long long ga = (unsigned long long)(uintptr_t)gptr;
  v4u g0;
  g0[0] = 1u;                                   // count=1, user descriptor
  g0[1] = lds_byte_off;                         // lds_addr
  g0[2] = (unsigned int)(ga & 0xffffffffu);     // global_addr[31:0]
  g0[3] = (unsigned int)((ga >> 32) & 0x1ffffffu) | (2u << 30); // [56:32] | type=2

  v8i g1;
  g1[0] = (int)((2u << 16)       // data_size = 2 -> 4 bytes
              | (1u << 20)       // pad_enable
              | (6u << 22)       // pad_interval: 1<<6 8-byte units = 128 DWORDs
              | (3u << 25));     // pad_amount: code 3 = 4 DWORDs
  g1[1] = (int)(128u << 16);     // tensor_dim0[15:0] = 128 (bits 63:48)
  g1[2] = (int)(64u  << 16);     // tensor_dim0 hi = 0 | tensor_dim1[15:0] = 64
  g1[3] = (int)(128u << 16);     // tensor_dim1 hi = 0 | tile_dim0 = 128
  g1[4] = (int)(64u);            // tile_dim1 = 64 | tile_dim2 = 0
  g1[5] = (int)(128u);           // tensor_dim0_stride[31:0] = 128 elements
  g1[6] = 0;                     // stride hi | tensor_dim1_stride lo = 0
  g1[7] = 0;

  v4i gz4 = {0, 0, 0, 0};        // groups 2/3 unused (2-D tensor)
  v8i gz8 = {0, 0, 0, 0, 0, 0, 0, 0};
  // 6-arg form (this toolchain): (g0, g1, g2, g3, g4, cpol)
  __builtin_amdgcn_tensor_load_to_lds(g0, g1, gz4, gz4, gz8, 0);
#endif
}

// ---------------- streaming attention (memory-bound core) ----------------
// One block per batch element. TDM stages k0 = [seq|seq_e|seq_t] (3 tiles of
// 64x128 fp32) into LDS (read-once from HBM, 402MB total ~ 17us at 23.3TB/s),
// then computes masked scores s[h,n] = k0_n . qk_h, softmax (writes attn_sq),
// and a0[h,:] = attn^T k0 (bf16 to ws).
__global__ __launch_bounds__(256) void attn_stream_kernel(
    const float* __restrict__ seq, const float* __restrict__ seq_e,
    const float* __restrict__ seq_t, const unsigned char* __restrict__ mask,
    const float* __restrict__ qk, __bf16* __restrict__ a0,
    float* __restrict__ attn_out)
{
  const int b = blockIdx.x;
  const int t = threadIdx.x;

  extern __shared__ float smem[];
  float* tile0 = smem;                      // 64*132 fp32 (seq)
  float* tile1 = tile0 + 64 * LDSROW;       // (seq_e)
  float* tile2 = tile1 + 64 * LDSROW;       // (seq_t)
  float* qks   = tile2 + 64 * LDSROW;       // 768
  float* sc    = qks + HH * MM;             // 128
  float* pr    = sc + 128;                  // 128
  float* mx    = pr + 128;                  // 2
  float* sm    = mx + 2;                    // 2
  float* tiles[3] = { tile0, tile1, tile2 };
  const float* segs[3] = { seq, seq_e, seq_t };

#if __has_builtin(__builtin_amdgcn_tensor_load_to_lds)
  // Wave 0 issues the three async tensor DMAs; TENSORcnt tracks completion.
  if (t < 32) {
    #pragma unroll
    for (int s = 0; s < 3; ++s)
      tdm_load_tile_64x128(segs[s] + (size_t)b * (NN * DD),
                           (unsigned int)(uintptr_t)tiles[s]);
    __builtin_amdgcn_s_wait_tensorcnt(0);
  }
#else
  // Fallback: cooperative vector loads + scalar LDS stores (padded rows).
  for (int s = 0; s < 3; ++s) {
    const float4* srcp = (const float4*)(segs[s] + (size_t)b * (NN * DD));
    for (int i = 0; i < 8; ++i) {
      int idx = t + i * 256;
      float4 vv = srcp[idx];
      int row = idx >> 5, col = (idx & 31) << 2;
      float* dst = tiles[s] + row * LDSROW + col;
      dst[0] = vv.x; dst[1] = vv.y; dst[2] = vv.z; dst[3] = vv.w;
    }
  }
#endif
  for (int i = t; i < HH * MM; i += 256) qks[i] = qk[(size_t)b * (HH * MM) + i];
  __syncthreads();   // publishes TDM-written LDS to all waves

  // scores: s[h,n] = sum_m k0[n,m] * qk[h,m]
  if (t < 128) {
    int h = t >> 6, n = t & 63;
    const float* qh = &qks[h * MM];
    float s = 0.f;
    #pragma unroll
    for (int g = 0; g < 3; ++g) {
      const float* kr = tiles[g] + n * LDSROW;
      for (int m = 0; m < 128; ++m) s += kr[m] * qh[g * 128 + m];
    }
    s *= 0.07216878364870323f;        // 1/sqrt(DK=192)
    if (mask[(size_t)b * NN + n]) s = -1e10f;
    sc[t] = s;
  }
  __syncthreads();
  if (t < 2) { float m = -3.4e38f; for (int n = 0; n < NN; ++n) m = fmaxf(m, sc[t * 64 + n]); mx[t] = m; }
  __syncthreads();
  if (t < 128) pr[t] = __expf(sc[t] - mx[t >> 6]);
  __syncthreads();
  if (t < 2) { float s = 0.f; for (int n = 0; n < NN; ++n) s += pr[t * 64 + n]; sm[t] = s; }
  __syncthreads();
  if (t < 128) {
    float a = pr[t] / sm[t >> 6];
    pr[t] = a;
    int h = t >> 6, n = t & 63;
    attn_out[((size_t)h * BB + b) * NN + n] = a;   // attn_sq[(h*B+b), n]
  }
  __syncthreads();

  // a0[h, m] = sum_n attn[h,n] * k0[n, m]   (768 outputs, bf16 to ws)
  #pragma unroll
  for (int i = 0; i < 3; ++i) {
    int j = t + i * 256;              // 0..767
    int h = j / MM, m = j - h * MM;
    const float* tg = tiles[m >> 7];
    int mc = m & 127;
    float acc = 0.f;
    for (int n = 0; n < NN; ++n) acc += pr[h * 64 + n] * tg[n * LDSROW + mc];
    a0[(size_t)b * (HH * MM) + j] = (__bf16)acc;
  }
}

// ---------------- residual + layernorm + concat ----------------
// x[b] = [ LN(res[b] + q0[b]) * g + beta  |  src[b] ]  (bf16, 512 wide)
__global__ __launch_bounds__(256) void ln_concat_kernel(
    const float* __restrict__ res, const float* __restrict__ src,
    const float* __restrict__ src_t, const float* __restrict__ g,
    const float* __restrict__ beta, __bf16* __restrict__ x)
{
  const int b = blockIdx.x;
  const int t = threadIdx.x;
  __shared__ float buf[MM];
  __shared__ float red[256];
  __shared__ float stats[2];

  float lsum = 0.f, lsq = 0.f;
  for (int j = t; j < MM; j += 256) {
    float q0 = (j < 128) ? src[(size_t)b * 128 + j]
             : (j >= 256 ? src_t[(size_t)b * 128 + (j - 256)] : 0.f);
    float v = res[(size_t)b * MM + j] + q0;
    buf[j] = v; lsum += v; lsq += v * v;
  }
  red[t] = lsum; __syncthreads();
  for (int s = 128; s > 0; s >>= 1) { if (t < s) red[t] += red[t + s]; __syncthreads(); }
  if (t == 0) stats[0] = red[0] * (1.0f / MM);
  __syncthreads();
  red[t] = lsq; __syncthreads();
  for (int s = 128; s > 0; s >>= 1) { if (t < s) red[t] += red[t + s]; __syncthreads(); }
  if (t == 0) {
    float mu = stats[0];
    float var = red[0] * (1.0f / MM) - mu * mu;
    stats[1] = rsqrtf(var + 1e-5f);
  }
  __syncthreads();
  const float mu = stats[0], rstd = stats[1];
  for (int j = t; j < MM; j += 256)
    x[(size_t)b * 512 + j] = (__bf16)((buf[j] - mu) * rstd * g[j] + beta[j]);
  if (t < 128)
    x[(size_t)b * 512 + MM + t] = (__bf16)src[(size_t)b * 128 + t];
}

// ---------------- launch ----------------

extern "C" void kernel_launch(void* const* d_in, const int* in_sizes, int n_in,
                              void* d_out, int out_size, void* d_ws, size_t ws_size,
                              hipStream_t stream) {
  // inputs (setup_inputs order)
  const float* src   = (const float*)d_in[0];
  const float* src_t = (const float*)d_in[1];
  const float* seq   = (const float*)d_in[2];
  const float* seq_t = (const float*)d_in[3];
  const float* seq_e = (const float*)d_in[4];
  const unsigned char* mask = (const unsigned char*)d_in[5];
  const float* Wq = (const float*)d_in[6];
  const float* Wk = (const float*)d_in[7];
  const float* Wv = (const float*)d_in[8];
  const float* Wo = (const float*)d_in[9];
  const float* bo = (const float*)d_in[10];
  const float* ln_g = (const float*)d_in[11];
  const float* ln_b = (const float*)d_in[12];
  const float* W1 = (const float*)d_in[13];
  const float* b1 = (const float*)d_in[14];
  const float* W2 = (const float*)d_in[15];
  const float* b2 = (const float*)d_in[16];

  float* y_out    = (float*)d_out;                    // (B,128)
  float* attn_out = (float*)d_out + (size_t)BB * DD;  // (H*B, N)

  // workspace carve-up (256B aligned)
  char* p = (char*)d_ws;
  auto alloc = [&](size_t bytes) -> void* {
    void* r = (void*)p; p += (bytes + 255) & ~(size_t)255; return r;
  };
  __bf16* Wqb  = (__bf16*)alloc((size_t)MM * MM * 2);
  __bf16* WkTb = (__bf16*)alloc((size_t)MM * MM * 2);      // Wk^T
  __bf16* Wvb  = (__bf16*)alloc((size_t)MM * MM * 2);
  __bf16* Wob  = (__bf16*)alloc((size_t)MM * MM * 2);
  __bf16* W1b  = (__bf16*)alloc((size_t)DD * 512 * 2);
  __bf16* W2b  = (__bf16*)alloc((size_t)DD * DD * 2);
  __bf16* Q0b  = (__bf16*)alloc((size_t)BB * MM * 2);
  __bf16* qb   = (__bf16*)alloc((size_t)BB * MM * 2);
  float*  qkf  = (float*) alloc((size_t)BB * HH * MM * 4);
  __bf16* a0b  = (__bf16*)alloc((size_t)BB * HH * MM * 2);
  __bf16* outvb= (__bf16*)alloc((size_t)BB * MM * 2);
  float*  resf = (float*) alloc((size_t)BB * MM * 4);
  __bf16* xb   = (__bf16*)alloc((size_t)BB * 512 * 2);
  __bf16* hb   = (__bf16*)alloc((size_t)BB * DD * 2);
  (void)ws_size; (void)n_in; (void)in_sizes; (void)out_size;

  // 1) weight prep (bf16; L2-resident)
  cvt_bf16_kernel<<<(MM*MM+255)/256, 256, 0, stream>>>(Wq, Wqb, MM*MM);
  cvt_bf16_kernel<<<(MM*MM+255)/256, 256, 0, stream>>>(Wv, Wvb, MM*MM);
  cvt_bf16_kernel<<<(MM*MM+255)/256, 256, 0, stream>>>(Wo, Wob, MM*MM);
  cvt_bf16_kernel<<<(DD*512+255)/256, 256, 0, stream>>>(W1, W1b, DD*512);
  cvt_bf16_kernel<<<(DD*DD+255)/256, 256, 0, stream>>>(W2, W2b, DD*DD);
  transpose384_kernel<<<(MM*MM+255)/256, 256, 0, stream>>>(Wk, WkTb);
  build_q0_kernel<<<(BB*MM+255)/256, 256, 0, stream>>>(src, src_t, Q0b);

  // 2) q = Q0 @ Wq^T            (B x 384), bf16
  gemm_bf16_kernel<<<dim3(MM/64, BB/64), 256, 0, stream>>>(
      Q0b, MM, Wqb, MM, nullptr, qb, MM, nullptr, MM, 0);

  // 3) qk_h = q_h @ WkT_h^T     (B x 384 per head), f32
  for (int h = 0; h < HH; ++h)
    gemm_bf16_kernel<<<dim3(MM/64, BB/64), 256, 0, stream>>>(
        qb + h * DK, MM, WkTb + h * DK, MM,
        qkf + h * MM, nullptr, HH * MM, nullptr, DK, 0);

  // 4) streaming attention core (TDM-staged, read-once 402MB)
  const size_t attn_lds = (size_t)(3 * 64 * LDSROW + HH * MM + 128 + 128 + 4)
                          * sizeof(float);
  attn_stream_kernel<<<BB, 256, attn_lds, stream>>>(seq, seq_e, seq_t, mask, qkf,
                                                    a0b, attn_out);

  // 5) out_v[:, h*192:(h+1)*192] = a0_h @ Wv_h^T   (bf16)
  for (int h = 0; h < HH; ++h)
    gemm_bf16_kernel<<<dim3(DK/64, BB/64), 256, 0, stream>>>(
        a0b + h * MM, HH * MM, Wvb + (size_t)h * DK * MM, MM,
        nullptr, outvb + h * DK, MM, nullptr, MM, 0);

  // 6) res = out_v @ Wo^T + bo  (f32)
  gemm_bf16_kernel<<<dim3(MM/64, BB/64), 256, 0, stream>>>(
      outvb, MM, Wob, MM, resf, nullptr, MM, bo, MM, 0);

  // 7) residual + layernorm + concat -> x (B x 512 bf16)
  ln_concat_kernel<<<BB, 256, 0, stream>>>(resf, src, src_t, ln_g, ln_b, xb);

  // 8) h = relu(x @ W1^T + b1)  (bf16)
  gemm_bf16_kernel<<<dim3(DD/64, BB/64), 256, 0, stream>>>(
      xb, 512, W1b, 512, nullptr, hb, DD, b1, 512, 1);

  // 9) y = h @ W2^T + b2        (f32, straight to d_out)
  gemm_bf16_kernel<<<dim3(DD/64, BB/64), 256, 0, stream>>>(
      hb, DD, W2b, DD, y_out, nullptr, DD, b2, DD, 0);
}
